// SlotRNN_53944789238070
// MI455X (gfx1250) — matrix operand order repaired and checked by reading, not compiled, not run on hardware
//
#include <hip/hip_runtime.h>

// ---------------------------------------------------------------------------
// MI455X (gfx1250) 2-layer GRU + linear + log_softmax(T)
//   - bf16 V_WMMA_F32_16X16X32_BF16 for all GEMMs (M=16 tile == batch size)
//   - persistent-grid recurrent scan, 1 atomic grid barrier per timestep,
//     ping-pong bf16 hidden state, gate math fused into the WMMA waves
//   - k-loops kept partially rolled so W_hh streams from L2 every step
//     (prevents LICM from hoisting B-fragments into scratch spills)
//   - fast v_exp_f32/v_rcp_f32 sigmoid+tanh to shorten the per-step serial tail
// Workspace requirement: ~257 MB.
// ---------------------------------------------------------------------------

typedef __bf16 bf16;
typedef __attribute__((ext_vector_type(16))) __bf16 v16bf;
typedef __attribute__((ext_vector_type(8)))  __bf16 v8bf;
typedef __attribute__((ext_vector_type(8)))  float  v8f;

#define HD  1024
#define TT  1024
#define BB  16
#define G3  3072
#define BT  (BB*TT)

#define LOG2E 1.4426950408889634f

// fast sigmoid: v_exp_f32 (base-2) + v_rcp_f32; saturates correctly at +-inf
__device__ __forceinline__ float fast_sigmoid(float x) {
  return __builtin_amdgcn_rcpf(1.0f + __builtin_amdgcn_exp2f(-x * LOG2E));
}
// fast tanh: 1 - 2/(1+exp(2x)); exp2 overflow -> rcp(inf)=0 -> +-1 (graceful)
__device__ __forceinline__ float fast_tanh(float x) {
  return 1.0f - 2.0f * __builtin_amdgcn_rcpf(1.0f + __builtin_amdgcn_exp2f(x * (2.0f * LOG2E)));
}

// ---------------- fp32 -> bf16 weight conversion ---------------------------
__global__ void k_f2bf(const float* __restrict__ src, bf16* __restrict__ dst, int n) {
  int i = blockIdx.x * blockDim.x + threadIdx.x;
  int stride = gridDim.x * blockDim.x;
  for (; i < n; i += stride) dst[i] = (bf16)src[i];
}

// ---------------- grid-wide barrier (monotonic counter) --------------------
__device__ __forceinline__ void grid_sync(unsigned* bar, unsigned nwg, unsigned& gen) {
  __syncthreads();
  gen += nwg;
  if (threadIdx.x == 0) {
    __threadfence();
    __hip_atomic_fetch_add(bar, 1u, __ATOMIC_ACQ_REL, __HIP_MEMORY_SCOPE_AGENT);
    while (__hip_atomic_load(bar, __ATOMIC_ACQUIRE, __HIP_MEMORY_SCOPE_AGENT) < gen) {
      __builtin_amdgcn_s_sleep(2);
    }
    __threadfence();
  }
  __syncthreads();
}

// ---------------- generic 16-row WMMA GEMM ---------------------------------
// Computes out[m0+mm, n] = sum_k A[m0+mm, k] * W[n, k] + bias[n]   (fp32 out)
// A rows staged in LDS as bf16 (gathered from emb via tokens, or direct bf16).
// Each wave: 16x64 output strip (4 accumulators sharing one A fragment).
template<bool GATHER>
__global__ __launch_bounds__(256)
void k_gemm16(const float* __restrict__ emb, const int* __restrict__ tokens,
              const bf16* __restrict__ Abf,
              const bf16* __restrict__ W, const float* __restrict__ bias,
              float* __restrict__ out, int ncols) {
  __shared__ bf16 As[16 * HD];                       // 32 KB of the 320 KB WGP LDS
  const int tid = threadIdx.x;
  const int m0  = blockIdx.x * 16;

  // stage 16 x 1024 A tile into LDS (bf16), 8 elements per thread per iter
  for (int i8 = tid; i8 < (16 * HD) / 8; i8 += 256) {
    const int r = i8 >> 7;
    const int c = (i8 & 127) << 3;
    v8bf v;
    if (GATHER) {
      const float* src = emb + (size_t)tokens[m0 + r] * HD + c;
      #pragma unroll
      for (int u = 0; u < 8; ++u) v[u] = (bf16)src[u];
    } else {
      v = *(const v8bf*)(Abf + (size_t)(m0 + r) * HD + c);
    }
    *(v8bf*)(&As[r * HD + c]) = v;                   // ds_store_b128
  }
  __syncthreads();

  const int wave = tid >> 5;
  const int lane = tid & 31;
  const int col0 = blockIdx.y * 512 + wave * 64;
  if (col0 >= ncols) return;
  const int m  = lane & 15;                          // A row / B column lane
  const int hi = lane >> 4;                          // half-wave select

  v8f acc0 = {}, acc1 = {}, acc2 = {}, acc3 = {};
  #pragma unroll 4
  for (int k = 0; k < HD; k += 32) {
    // A 16x32 fragment: lane m holds row m; VGPR0-3 K=kb..kb+7, VGPR4-7 K=kb+8..kb+15
    const int kb = k + hi * 16;
    union { v16bf v; v8bf h[2]; } a;
    a.h[0] = *(const v8bf*)(&As[m * HD + kb]);       // ds_load_b128
    a.h[1] = *(const v8bf*)(&As[m * HD + kb + 8]);
    // B 32x16 fragment: lane = column col0+16j+m, 16 contiguous K values
    const bf16* wp = W + (size_t)(col0 + m) * HD + k + hi * 16;
    v16bf b0 = *(const v16bf*)(wp);
    v16bf b1 = *(const v16bf*)(wp + 16 * HD);
    v16bf b2 = *(const v16bf*)(wp + 32 * HD);
    v16bf b3 = *(const v16bf*)(wp + 48 * HD);
    acc0 = __builtin_amdgcn_wmma_f32_16x16x32_bf16(false, a.v, false, b0, (short)0, acc0, false, false);
    acc1 = __builtin_amdgcn_wmma_f32_16x16x32_bf16(false, a.v, false, b1, (short)0, acc1, false, false);
    acc2 = __builtin_amdgcn_wmma_f32_16x16x32_bf16(false, a.v, false, b2, (short)0, acc2, false, false);
    acc3 = __builtin_amdgcn_wmma_f32_16x16x32_bf16(false, a.v, false, b3, (short)0, acc3, false, false);
  }
  // C/D layout: VGPR r, lane L -> M = r + 8*(L>=16), N = L&15
  v8f accs[4] = {acc0, acc1, acc2, acc3};
  #pragma unroll
  for (int j = 0; j < 4; ++j) {
    const int n  = col0 + j * 16 + m;
    const float bs = bias[n];
    #pragma unroll
    for (int r = 0; r < 8; ++r) {
      const int mm = r + hi * 8;
      out[(size_t)(m0 + mm) * ncols + n] = accs[j][r] + bs;
    }
  }
}

// ---------------- persistent recurrent GRU scan ----------------------------
// 8 blocks x 8 waves = 64 waves; wave gw owns hidden columns j0..j0+15 for the
// full batch.  Per step: stage h (bf16, ping-pong) into LDS, 3 WMMA chains
// (r/z/n gates) sharing the A fragment, gate math fully wave-local.
// The k-loop is kept partially rolled (#pragma unroll 2): its W_hh addresses
// then vary with k inside the loop, so the compiler cannot hoist the
// (t-invariant) B-fragments out of the t-loop and spill them to scratch --
// W_hh must stream from the 192 MB L2 every step, which is the fast path.
__global__ __launch_bounds__(256)
void k_gru_scan(const bf16* __restrict__ Whh, const float* __restrict__ bhh,
                const float* __restrict__ gx,   // [B*T, 3H] fp32, row = b*T + t
                bf16* __restrict__ hb,          // 2 x [B, H] bf16 ping-pong state
                bf16* __restrict__ hout,        // [B, T, H] bf16 layer output
                unsigned* __restrict__ bar) {
  __shared__ bf16 Hs[BB * HD];                      // 32 KB
  const int tid  = threadIdx.x;
  const int wave = tid >> 5;
  const int lane = tid & 31;
  const int gw   = blockIdx.x * 8 + wave;          // 0..63
  const int j0   = gw * 16;
  const int m    = lane & 15;
  const int hi   = lane >> 4;
  const int j    = j0 + m;
  const unsigned NWG = gridDim.x;
  unsigned gen = 0;

  // zero ping buffer 0 (read at t=0); buffer 1 is written before first read
  for (int i = blockIdx.x * 256 + tid; i < BB * HD; i += NWG * 256) hb[i] = (bf16)0.0f;

  float hcur[8];
  #pragma unroll
  for (int r = 0; r < 8; ++r) hcur[r] = 0.0f;

  const float b_r = bhh[j];
  const float b_z = bhh[HD + j];
  const float b_n = bhh[2 * HD + j];

  grid_sync(bar, NWG, gen);                         // zeros visible everywhere

  for (int t = 0; t < TT; ++t) {
    const bf16* hread = hb + (size_t)(t & 1) * (BB * HD);
    bf16* hwrite      = hb + (size_t)((t & 1) ^ 1) * (BB * HD);

    // stage previous h into LDS
    for (int i8 = tid; i8 < (BB * HD) / 8; i8 += 256)
      *(v8bf*)(&Hs[i8 * 8]) = *(const v8bf*)(hread + (size_t)i8 * 8);
    __syncthreads();

    v8f ar = {}, az = {}, an = {};
    #pragma unroll 2
    for (int k = 0; k < HD; k += 32) {
      const int kb = k + hi * 16;
      union { v16bf v; v8bf h[2]; } a;
      a.h[0] = *(const v8bf*)(&Hs[m * HD + kb]);
      a.h[1] = *(const v8bf*)(&Hs[m * HD + kb + 8]);
      const bf16* wp = Whh + k + hi * 16;
      v16bf br_ = *(const v16bf*)(wp + (size_t)(j) * HD);
      v16bf bz_ = *(const v16bf*)(wp + (size_t)(HD + j) * HD);
      v16bf bn_ = *(const v16bf*)(wp + (size_t)(2 * HD + j) * HD);
      ar = __builtin_amdgcn_wmma_f32_16x16x32_bf16(false, a.v, false, br_, (short)0, ar, false, false);
      az = __builtin_amdgcn_wmma_f32_16x16x32_bf16(false, a.v, false, bz_, (short)0, az, false, false);
      an = __builtin_amdgcn_wmma_f32_16x16x32_bf16(false, a.v, false, bn_, (short)0, an, false, false);
    }

    // gates: element (r, lane) -> batch b_ = r + 8*hi, hidden col j
    #pragma unroll
    for (int r = 0; r < 8; ++r) {
      const int b_ = r + hi * 8;
      const float* gp = gx + ((size_t)b_ * TT + t) * G3 + j;
      const float xr = __builtin_nontemporal_load(gp);            // read-once data:
      const float xz = __builtin_nontemporal_load(gp + HD);       // keep W_hh resident
      const float xn = __builtin_nontemporal_load(gp + 2 * HD);   // in L2 instead
      const float rr = fast_sigmoid(xr + ar[r] + b_r);
      const float zz = fast_sigmoid(xz + az[r] + b_z);
      const float nn = fast_tanh   (xn + rr * (an[r] + b_n));
      const float hn = (1.0f - zz) * nn + zz * hcur[r];
      hcur[r] = hn;
      const bf16 hv = (bf16)hn;
      hwrite[b_ * HD + j] = hv;
      hout[((size_t)b_ * TT + t) * HD + j] = hv;
    }
    grid_sync(bar, NWG, gen);                       // h_t visible for t+1 staging
  }
}

// ---------------- log_softmax over T axis ----------------------------------
// out[b,t,c] = logit[b,t,c] - logsumexp_t(logit[b,:,c]); threads = c (coalesced)
__global__ __launch_bounds__(128)
void k_logsoftmax_T(const float* __restrict__ logits, float* __restrict__ out) {
  const int b = blockIdx.x;
  const int c = threadIdx.x;
  const float* p = logits + (size_t)b * TT * 128 + c;
  float mx = -3.402823466e38f;
  for (int t = 0; t < TT; ++t) mx = fmaxf(mx, p[(size_t)t * 128]);
  float s = 0.0f;
  for (int t = 0; t < TT; ++t) s += __builtin_amdgcn_exp2f((p[(size_t)t * 128] - mx) * LOG2E);
  const float lse = mx + __logf(s);
  float* q = out + (size_t)b * TT * 128 + c;
  for (int t = 0; t < TT; ++t) q[(size_t)t * 128] = p[(size_t)t * 128] - lse;
}

// ---------------------------------------------------------------------------
extern "C" void kernel_launch(void* const* d_in, const int* in_sizes, int n_in,
                              void* d_out, int out_size, void* d_ws, size_t ws_size,
                              hipStream_t stream) {
  const int*   tokens = (const int*)  d_in[0];
  const float* emb    = (const float*)d_in[1];
  const float* W_ih0  = (const float*)d_in[2];
  const float* W_hh0  = (const float*)d_in[3];
  const float* b_ih0  = (const float*)d_in[4];
  const float* b_hh0  = (const float*)d_in[5];
  const float* W_ih1  = (const float*)d_in[6];
  const float* W_hh1  = (const float*)d_in[7];
  const float* b_ih1  = (const float*)d_in[8];
  const float* b_hh1  = (const float*)d_in[9];
  const float* W_lin  = (const float*)d_in[10];
  const float* b_lin  = (const float*)d_in[11];
  float* out = (float*)d_out;

  char* w = (char*)d_ws;
  auto take = [&](size_t bytes) -> char* {
    char* p = w; w += (bytes + 255) & ~(size_t)255; return p;
  };
  bf16*  Wb_ih0 = (bf16*) take((size_t)G3 * HD * 2);
  bf16*  Wb_hh0 = (bf16*) take((size_t)G3 * HD * 2);
  bf16*  Wb_ih1 = (bf16*) take((size_t)G3 * HD * 2);
  bf16*  Wb_hh1 = (bf16*) take((size_t)G3 * HD * 2);
  bf16*  Wb_lin = (bf16*) take((size_t)128 * HD * 2);
  bf16*  hb     = (bf16*) take((size_t)2 * BB * HD * 2);   // ping-pong state
  bf16*  hseq   = (bf16*) take((size_t)BT * HD * 2);       // layer output (reused)
  float* gx     = (float*)take((size_t)BT * G3 * 4);       // 192 MB (reused per layer)
  float* logits = (float*)take((size_t)BT * 128 * 4);
  unsigned* bar = (unsigned*)take(256);

  // 1. weights -> bf16
  k_f2bf<<<256, 256, 0, stream>>>(W_ih0, Wb_ih0, G3 * HD);
  k_f2bf<<<256, 256, 0, stream>>>(W_hh0, Wb_hh0, G3 * HD);
  k_f2bf<<<256, 256, 0, stream>>>(W_ih1, Wb_ih1, G3 * HD);
  k_f2bf<<<256, 256, 0, stream>>>(W_hh1, Wb_hh1, G3 * HD);
  k_f2bf<<<256, 256, 0, stream>>>(W_lin, Wb_lin, 128 * HD);

  // 2. layer-0 input gates: gx = emb[tokens] @ W_ih0^T + b_ih0   (fused gather)
  k_gemm16<true><<<dim3(BT / 16, 6), 256, 0, stream>>>(
      emb, tokens, nullptr, Wb_ih0, b_ih0, gx, G3);

  // 3. layer-0 recurrent scan -> hseq = h1
  hipMemsetAsync(bar, 0, 256, stream);
  k_gru_scan<<<8, 256, 0, stream>>>(Wb_hh0, b_hh0, gx, hb, hseq, bar);

  // 4. layer-1 input gates: gx = h1 @ W_ih1^T + b_ih1
  k_gemm16<false><<<dim3(BT / 16, 6), 256, 0, stream>>>(
      nullptr, nullptr, hseq, Wb_ih1, b_ih1, gx, G3);

  // 5. layer-1 recurrent scan -> hseq = h2 (overwrites h1, no longer needed)
  hipMemsetAsync(bar, 0, 256, stream);
  k_gru_scan<<<8, 256, 0, stream>>>(Wb_hh1, b_hh1, gx, hb, hseq, bar);

  // 6. final linear: logits = h2 @ W_lin^T + b_lin
  k_gemm16<false><<<dim3(BT / 16, 1), 256, 0, stream>>>(
      nullptr, nullptr, hseq, Wb_lin, b_lin, logits, 128);

  // 7. log_softmax over the T axis
  k_logsoftmax_T<<<BB, 128, 0, stream>>>(logits, out);
}
